// TransformerEncoder_55207509623349
// MI455X (gfx1250) — matrix operand order repaired and checked by reading, not compiled, not run on hardware
//
#include <hip/hip_runtime.h>

typedef __bf16 bf16_t;
typedef bf16_t bf16x16 __attribute__((ext_vector_type(16)));
typedef bf16_t bf16x8  __attribute__((ext_vector_type(8)));
typedef float  f32x8   __attribute__((ext_vector_type(8)));
typedef int    i32x4   __attribute__((ext_vector_type(4)));

#if defined(__has_builtin)
#if __has_builtin(__builtin_amdgcn_global_load_async_to_lds_b128)
#define HAVE_ASYNC_LDS 1
#endif
#if __has_builtin(__builtin_amdgcn_s_wait_asynccnt)
#define HAVE_WAIT_ASYNC 1
#endif
#endif

#if HAVE_ASYNC_LDS
typedef __attribute__((address_space(1))) i32x4 as1_i32x4;
typedef __attribute__((address_space(3))) i32x4 as3_i32x4;
#endif

// A-fragment K index for 16-bit WMMA A (16x32), per ISA 7.12.2:
// lanes 0-15: vgpr0..3 -> K=0..7, vgpr4..7 -> K=16..23 ; lanes 16-31: +8
__device__ __forceinline__ int a_kidx(int j, int g) {
    return (j < 4 ? 2 * j : 16 + 2 * (j - 4)) + 8 * g;
}

// ---------------------------------------------------------------------------
// GEMM: C[M,N] = A[M,K](bf16) * B[K,N](f32, cvt to bf16 on the fly)
// Block = 256 threads = 8 waves. Block tile 128(M) x 64(N); each wave owns a
// 16x64 strip (4 C tiles, one A fragment reused across 4 WMMAs per k-step).
// B k-slice (32x64 f32, 8KB) is staged in LDS once per block and shared by
// all 8 waves; staged via gfx1250 async-copy-to-LDS when available.
// M,N,K multiples of 128/64/32 at all call sites -> no bounds checks, EXEC
// stays all-ones for WMMA.
// ---------------------------------------------------------------------------
template <bool BIAS, bool RES, bool RELU, bool OUTF, bool OUTB>
__global__ void __launch_bounds__(256)
gemm_wmma(const bf16_t* __restrict__ A, const float* __restrict__ Bw,
          const float* __restrict__ bias, const float* __restrict__ resid,
          float* __restrict__ Cf, bf16_t* __restrict__ Cb,
          int M, int N, int K)
{
    __shared__ float ldsBf[32][64];   // [K-slice][N-slice]

    const int wave = threadIdx.x >> 5;
    const int lane = threadIdx.x & 31;
    const int g  = lane >> 4;         // half-wave
    const int rc = lane & 15;         // A row / B,C col within 16x16 tile

    const int nBlk = N >> 6;
    const int bm = blockIdx.x / nBlk;
    const int bn = blockIdx.x % nBlk;
    const int n0 = bn * 64;
    const int row0 = bm * 128 + wave * 16;

    const bf16_t* Arow = A + (size_t)(row0 + rc) * K;

    f32x8 acc[4] = {f32x8{}, f32x8{}, f32x8{}, f32x8{}};

    for (int k0 = 0; k0 < K; k0 += 32) {
        __syncthreads();   // previous iteration's LDS reads complete
        // ---- stage B k-slice (32 rows x 64 cols f32) into LDS ----
#if HAVE_ASYNC_LDS
#pragma unroll
        for (int i = 0; i < 2; ++i) {
            const int q  = i * 256 + threadIdx.x;   // 0..511 float4 segments
            const int kr = q >> 4;                  // 16 float4 per row
            const int c4 = (q & 15) * 4;
            const float* gp = Bw + (size_t)(k0 + kr) * N + n0 + c4;
            __builtin_amdgcn_global_load_async_to_lds_b128(
                (as1_i32x4*)gp, (as3_i32x4*)&ldsBf[kr][c4], 0, 0);
        }
#if HAVE_WAIT_ASYNC
        __builtin_amdgcn_s_wait_asynccnt(0);
#else
        asm volatile("s_wait_asynccnt 0" ::: "memory");
#endif
#else
#pragma unroll
        for (int i = 0; i < 2; ++i) {
            const int q  = i * 256 + threadIdx.x;
            const int kr = q >> 4;
            const int c4 = (q & 15) * 4;
            const float4 v = *(const float4*)(Bw + (size_t)(k0 + kr) * N + n0 + c4);
            *(float4*)&ldsBf[kr][c4] = v;
        }
#endif
        __syncthreads();

        // ---- A fragment: two contiguous 16B loads + shuffle ----
        if (k0 + 32 < K) __builtin_prefetch(Arow + k0 + 32, 0, 3);
        const bf16x8 alo = *(const bf16x8*)(Arow + k0 + 8 * g);
        const bf16x8 ahi = *(const bf16x8*)(Arow + k0 + 16 + 8 * g);
        const bf16x16 a = __builtin_shufflevector(alo, ahi,
            0, 1, 2, 3, 4, 5, 6, 7, 8, 9, 10, 11, 12, 13, 14, 15);

        // ---- 4 B fragments from LDS, 4 WMMAs ----
#pragma unroll
        for (int nt = 0; nt < 4; ++nt) {
            bf16x16 b;
            const int col = nt * 16 + rc;
#pragma unroll
            for (int j = 0; j < 8; ++j) {
                const int kr = a_kidx(j, g);
                b[2 * j]     = (bf16_t)ldsBf[kr][col];
                b[2 * j + 1] = (bf16_t)ldsBf[kr + 1][col];
            }
            acc[nt] = __builtin_amdgcn_wmma_f32_16x16x32_bf16(
                false, a, false, b, (short)0, acc[nt], false, false);
        }
    }

    // ---- branch-free epilogue (template-specialized) ----
#pragma unroll
    for (int nt = 0; nt < 4; ++nt) {
        const int col = n0 + nt * 16 + rc;
#pragma unroll
        for (int r = 0; r < 8; ++r) {
            const int row = row0 + r + 8 * g;
            float v = acc[nt][r];
            if (BIAS) v += bias[col];
            if (RES)  v += resid[(size_t)row * N + col];
            if (RELU) v = v > 0.f ? v : 0.f;
            if (OUTF) Cf[(size_t)row * N + col] = v;
            if (OUTB) Cb[(size_t)row * N + col] = (bf16_t)v;
        }
    }
}

// ---------------------------------------------------------------------------
// Flash-attention (no mask): one workgroup = 4 waves = 64 query rows of one
// (batch, head). Streams key blocks of 32; scores and P*V via bf16 WMMA,
// online softmax (running max / sum) entirely in registers.
// ---------------------------------------------------------------------------
#define ATTN_HD 64
#define ATTN_NH 16

__global__ void __launch_bounds__(128)
flash_attn_wmma(const bf16_t* __restrict__ Q, const bf16_t* __restrict__ Km,
                const bf16_t* __restrict__ Vm, bf16_t* __restrict__ Om,
                int S, int Dm)
{
    __shared__ bf16_t ldsP[4][16][32];

    const int wave = threadIdx.x >> 5;
    const int lane = threadIdx.x & 31;
    const int g  = lane >> 4;
    const int rc = lane & 15;

    const int qblocks = S / 64;
    const int bh   = blockIdx.x / qblocks;
    const int qblk = blockIdx.x % qblocks;
    const int b    = bh / ATTN_NH;
    const int head = bh % ATTN_NH;
    const int q0   = qblk * 64 + wave * 16;

    const bf16_t* Qbase = Q  + (size_t)b * S * Dm + head * ATTN_HD;
    const bf16_t* Kbase = Km + (size_t)b * S * Dm + head * ATTN_HD;
    const bf16_t* Vbase = Vm + (size_t)b * S * Dm + head * ATTN_HD;

    // Q fragments (HD=64 -> two K=32 steps), loaded once via wide loads
    bf16x16 aq[2];
    {
        const bf16_t* qrow = Qbase + (size_t)(q0 + rc) * Dm;
        const bf16x8 q00 = *(const bf16x8*)(qrow + 8 * g);
        const bf16x8 q01 = *(const bf16x8*)(qrow + 16 + 8 * g);
        const bf16x8 q10 = *(const bf16x8*)(qrow + 32 + 8 * g);
        const bf16x8 q11 = *(const bf16x8*)(qrow + 48 + 8 * g);
        aq[0] = __builtin_shufflevector(q00, q01, 0,1,2,3,4,5,6,7,8,9,10,11,12,13,14,15);
        aq[1] = __builtin_shufflevector(q10, q11, 0,1,2,3,4,5,6,7,8,9,10,11,12,13,14,15);
    }

    f32x8 oacc[4] = {f32x8{}, f32x8{}, f32x8{}, f32x8{}};
    float m[8], l[8];
#pragma unroll
    for (int r = 0; r < 8; ++r) { m[r] = -3.0e38f; l[r] = 0.f; }

    for (int kpos = 0; kpos < S; kpos += 32) {
        // --- scores: 16 q-rows x 32 keys, K = HD = 64 ---
        f32x8 sc0 = {}, sc1 = {};
#pragma unroll
        for (int step = 0; step < 2; ++step) {
            bf16x16 bk0, bk1;
#pragma unroll
            for (int j = 0; j < 8; ++j) {
                const int hd = step * 32 + a_kidx(j, g);
                const bf16_t* k0p = Kbase + (size_t)(kpos + rc) * Dm + hd;
                const bf16_t* k1p = Kbase + (size_t)(kpos + 16 + rc) * Dm + hd;
                bk0[2 * j] = k0p[0]; bk0[2 * j + 1] = k0p[1];
                bk1[2 * j] = k1p[0]; bk1[2 * j + 1] = k1p[1];
            }
            sc0 = __builtin_amdgcn_wmma_f32_16x16x32_bf16(false, aq[step], false, bk0,
                                                          (short)0, sc0, false, false);
            sc1 = __builtin_amdgcn_wmma_f32_16x16x32_bf16(false, aq[step], false, bk1,
                                                          (short)0, sc1, false, false);
        }

        // --- online softmax update (scale 1/sqrt(64) = 0.125) ---
        float p0[8], p1[8], alpha[8];
#pragma unroll
        for (int r = 0; r < 8; ++r) {
            const float s0 = sc0[r] * 0.125f;
            const float s1 = sc1[r] * 0.125f;
            float mx = fmaxf(s0, s1);
#pragma unroll
            for (int off = 1; off < 16; off <<= 1)
                mx = fmaxf(mx, __shfl_xor(mx, off, 32));
            const float mn = fmaxf(m[r], mx);
            alpha[r] = __expf(m[r] - mn);
            p0[r] = __expf(s0 - mn);
            p1[r] = __expf(s1 - mn);
            float sum = p0[r] + p1[r];
#pragma unroll
            for (int off = 1; off < 16; off <<= 1)
                sum += __shfl_xor(sum, off, 32);
            l[r] = l[r] * alpha[r] + sum;
            m[r] = mn;
        }
#pragma unroll
        for (int r = 0; r < 8; ++r) {
            oacc[0][r] *= alpha[r]; oacc[1][r] *= alpha[r];
            oacc[2][r] *= alpha[r]; oacc[3][r] *= alpha[r];
        }

        // --- re-layout P: C-fragment -> LDS -> A-fragment ---
#pragma unroll
        for (int r = 0; r < 8; ++r) {
            ldsP[wave][r + 8 * g][rc]      = (bf16_t)p0[r];
            ldsP[wave][r + 8 * g][16 + rc] = (bf16_t)p1[r];
        }
        __syncthreads();
        bf16x16 ap;
        {
            const bf16_t* prow = &ldsP[wave][rc][0];
            const bf16x8 plo = *(const bf16x8*)(prow + 8 * g);
            const bf16x8 phi = *(const bf16x8*)(prow + 16 + 8 * g);
            ap = __builtin_shufflevector(plo, phi, 0,1,2,3,4,5,6,7,8,9,10,11,12,13,14,15);
        }
        __syncthreads();

        // --- O += P(16x32) @ V(32x64), 4 N-tiles of 16 ---
#pragma unroll
        for (int n = 0; n < 4; ++n) {
            bf16x16 bv;
#pragma unroll
            for (int j = 0; j < 8; ++j) {
                const int kr = a_kidx(j, g);
                const bf16_t* vp = Vbase + (size_t)(kpos + kr) * Dm + n * 16 + rc;
                bv[2 * j]     = vp[0];
                bv[2 * j + 1] = vp[Dm];
            }
            oacc[n] = __builtin_amdgcn_wmma_f32_16x16x32_bf16(false, ap, false, bv,
                                                              (short)0, oacc[n], false, false);
        }
    }

    // --- normalize by row sum, store bf16 [B,S,H,HD] ---
    bf16_t* Obase = Om + (size_t)b * S * Dm + head * ATTN_HD;
#pragma unroll
    for (int r = 0; r < 8; ++r) {
        const float inv = 1.0f / l[r];
        const int row = q0 + r + 8 * g;
#pragma unroll
        for (int n = 0; n < 4; ++n)
            Obase[(size_t)row * Dm + n * 16 + rc] = (bf16_t)(oacc[n][r] * inv);
    }
}

// ---------------------------------------------------------------------------
// LayerNorm over last dim (one row per 256-thread block), dual f32+bf16 out.
// ---------------------------------------------------------------------------
__global__ void __launch_bounds__(256)
layernorm_kernel(const float* __restrict__ X, const float* __restrict__ gamma,
                 const float* __restrict__ beta, float* __restrict__ Yf,
                 bf16_t* __restrict__ Yb, int Dm)
{
    __shared__ float red[8];
    const int row = blockIdx.x;
    const float* x = X + (size_t)row * Dm;
    const int t = threadIdx.x;

    float s = 0.f;
    for (int i = t; i < Dm; i += 256) s += x[i];
#pragma unroll
    for (int off = 1; off < 32; off <<= 1) s += __shfl_xor(s, off, 32);
    if ((t & 31) == 0) red[t >> 5] = s;
    __syncthreads();
    float mean = 0.f;
#pragma unroll
    for (int i = 0; i < 8; ++i) mean += red[i];
    mean /= (float)Dm;
    __syncthreads();

    float v = 0.f;
    for (int i = t; i < Dm; i += 256) { const float d = x[i] - mean; v += d * d; }
#pragma unroll
    for (int off = 1; off < 32; off <<= 1) v += __shfl_xor(v, off, 32);
    if ((t & 31) == 0) red[t >> 5] = v;
    __syncthreads();
    float var = 0.f;
#pragma unroll
    for (int i = 0; i < 8; ++i) var += red[i];
    var /= (float)Dm;
    const float rstd = rsqrtf(var + 1e-5f);

    for (int i = t; i < Dm; i += 256) {
        const float y = (x[i] - mean) * rstd * gamma[i] + beta[i];
        Yf[(size_t)row * Dm + i] = y;
        Yb[(size_t)row * Dm + i] = (bf16_t)y;
    }
}

// ---------------------------------------------------------------------------
// Embedding * sqrt(D) + positional encoding (reference's nonstandard PE:
// even col c -> sin(pos * 10000^(-2c/D)), odd col c -> cos(pos * 10000^(-2c/D)))
// ---------------------------------------------------------------------------
__global__ void __launch_bounds__(256)
embed_kernel(const int* __restrict__ x, const float* __restrict__ emb,
             float* __restrict__ hf, bf16_t* __restrict__ hb, int S, int Dm)
{
    const size_t idx = (size_t)blockIdx.x * 256 + threadIdx.x;
    const int d = (int)(idx % Dm);
    const size_t bs = idx / Dm;
    const int s = (int)(bs % S);
    const int tok = x[bs];
    float val = emb[(size_t)tok * Dm + d] * 32.0f;   // sqrt(1024)
    const float inv = __powf(10000.f, -2.f * (float)d / (float)Dm);
    const float ang = (float)s * inv;
    val += (d & 1) ? __cosf(ang) : __sinf(ang);
    hf[idx] = val;
    hb[idx] = (bf16_t)val;
}

// ---------------------------------------------------------------------------
extern "C" void kernel_launch(void* const* d_in, const int* in_sizes, int n_in,
                              void* d_out, int out_size, void* d_ws, size_t ws_size,
                              hipStream_t stream)
{
    const int BN = 2, SL = 2048, DM = 1024, HN = 16, LN = 6;
    const int MR = BN * SL;       // 4096 rows
    const int DF = 4 * DM;        // 4096 FFN dim

    const int*   x     = (const int*)  d_in[0];
    const float* embed = (const float*)d_in[1];
    const float* Wq    = (const float*)d_in[2];
    const float* Wk    = (const float*)d_in[3];
    const float* Wv    = (const float*)d_in[4];
    const float* Wo    = (const float*)d_in[5];
    const float* bo    = (const float*)d_in[6];
    const float* ln1g  = (const float*)d_in[7];
    const float* ln1b  = (const float*)d_in[8];
    const float* W1    = (const float*)d_in[9];
    const float* b1    = (const float*)d_in[10];
    const float* W2    = (const float*)d_in[11];
    const float* b2    = (const float*)d_in[12];
    const float* ln2g  = (const float*)d_in[13];
    const float* ln2b  = (const float*)d_in[14];

    char* ws = (char*)d_ws;
    size_t off = 0;
    auto alloc = [&](size_t bytes) -> char* {
        char* p = ws + off;
        off += (bytes + 255) & ~(size_t)255;
        return p;
    };
    float*  hF   = (float*) alloc((size_t)MR * DM * 4);
    float*  tmpF = (float*) alloc((size_t)MR * DM * 4);
    float*  n1F  = (float*) alloc((size_t)MR * DM * 4);
    bf16_t* hB   = (bf16_t*)alloc((size_t)MR * DM * 2);
    bf16_t* n1B  = (bf16_t*)alloc((size_t)MR * DM * 2);
    bf16_t* qB   = (bf16_t*)alloc((size_t)MR * DM * 2);
    bf16_t* kB   = (bf16_t*)alloc((size_t)MR * DM * 2);
    bf16_t* vB   = (bf16_t*)alloc((size_t)MR * DM * 2);
    bf16_t* aB   = (bf16_t*)alloc((size_t)MR * DM * 2);
    bf16_t* fB   = (bf16_t*)alloc((size_t)MR * DF * 2);

    // h = embed[x]*sqrt(D) + PE
    embed_kernel<<<(unsigned)(((size_t)MR * DM) / 256), 256, 0, stream>>>(x, embed, hF, hB, SL, DM);

    const unsigned gDD = (unsigned)((MR / 128) * (DM / 64));   // 512 blocks
    const unsigned gDF = (unsigned)((MR / 128) * (DF / 64));   // 2048 blocks
    const unsigned gAT = (unsigned)(BN * HN * (SL / 64));      // 1024 blocks

    for (int l = 0; l < LN; ++l) {
        const float* wq   = Wq   + (size_t)l * DM * DM;
        const float* wk   = Wk   + (size_t)l * DM * DM;
        const float* wv   = Wv   + (size_t)l * DM * DM;
        const float* wo   = Wo   + (size_t)l * DM * DM;
        const float* bol  = bo   + (size_t)l * DM;
        const float* g1   = ln1g + (size_t)l * DM;
        const float* bb1  = ln1b + (size_t)l * DM;
        const float* w1   = W1   + (size_t)l * DM * DF;
        const float* bf1  = b1   + (size_t)l * DF;
        const float* w2   = W2   + (size_t)l * DF * DM;
        const float* bf2  = b2   + (size_t)l * DM;
        const float* g2   = ln2g + (size_t)l * DM;
        const float* bb2  = ln2b + (size_t)l * DM;

        // Q, K, V projections (no bias in reference) -> bf16
        gemm_wmma<false, false, false, false, true><<<gDD, 256, 0, stream>>>(
            hB, wq, nullptr, nullptr, nullptr, qB, MR, DM, DM);
        gemm_wmma<false, false, false, false, true><<<gDD, 256, 0, stream>>>(
            hB, wk, nullptr, nullptr, nullptr, kB, MR, DM, DM);
        gemm_wmma<false, false, false, false, true><<<gDD, 256, 0, stream>>>(
            hB, wv, nullptr, nullptr, nullptr, vB, MR, DM, DM);

        // streaming attention -> aB (bf16 [B,S,H,HD])
        flash_attn_wmma<<<gAT, 128, 0, stream>>>(qB, kB, vB, aB, SL, DM);

        // output proj + bias + residual(h) -> tmpF ; LN1 -> n1F / n1B
        gemm_wmma<true, true, false, true, false><<<gDD, 256, 0, stream>>>(
            aB, wo, bol, hF, tmpF, nullptr, MR, DM, DM);
        layernorm_kernel<<<(unsigned)MR, 256, 0, stream>>>(tmpF, g1, bb1, n1F, n1B, DM);

        // FFN: relu(n1 @ W1 + b1) -> fB ; fB @ W2 + b2 + n1 -> tmpF ; LN2 -> hF/hB
        gemm_wmma<true, false, true, false, true><<<gDF, 256, 0, stream>>>(
            n1B, w1, bf1, nullptr, nullptr, fB, MR, DF, DM);
        gemm_wmma<true, true, false, true, false><<<gDD, 256, 0, stream>>>(
            fB, w2, bf2, n1F, tmpF, nullptr, MR, DM, DF);
        layernorm_kernel<<<(unsigned)MR, 256, 0, stream>>>(tmpF, g2, bb2, hF, hB, DM);
    }

    (void)hipMemcpyAsync(d_out, hF, (size_t)MR * DM * sizeof(float),
                         hipMemcpyDeviceToDevice, stream);
}
